// GCNLayer_69672959476101
// MI455X (gfx1250) — matrix operand order repaired and checked by reading, not compiled, not run on hardware
//
#include <hip/hip_runtime.h>
#include <hip/hip_bf16.h>

#define NN 16384
#define FF 128
#define KC 128             // K-chunk staged in LDS
#define SP (KC + 8)        // padded LDS row stride (bf16 elems)
#define SP2 (FF + 8)
#define KSPLIT 4
#define KRANGE (NN / KSPLIT)

typedef __attribute__((ext_vector_type(16))) __bf16 bf16x16;
typedef __attribute__((ext_vector_type(8)))  float  f32x8;

// scheduling pin helper (no-op if builtin unavailable)
#if defined(__has_builtin)
#  if __has_builtin(__builtin_amdgcn_sched_group_barrier)
#    define SGB(m, s, i) __builtin_amdgcn_sched_group_barrier(m, s, i)
#  endif
#endif
#ifndef SGB
#  define SGB(m, s, i)
#endif

union BFrag {
    bf16x16 v;
    unsigned short u[16];
    uint4 q[2];
};

__device__ __forceinline__ unsigned short f2bf(float f) {
    unsigned int x = __float_as_uint(f);
    x += 0x7fffu + ((x >> 16) & 1u);          // round-to-nearest-even
    return (unsigned short)(x >> 16);
}

// ---------------- pass 1: degree -> norm = rsqrt(rowsum) ----------------
__global__ void k_rowsum(const float* __restrict__ A, float* __restrict__ norm) {
    __shared__ float red[256];
    const int row = blockIdx.x;
    const float4* ap = (const float4*)(A + (size_t)row * NN);
    float s = 0.f;
    #pragma unroll 4
    for (int i = 0; i < NN / 4; i += 256) {
        float4 v = ap[i + threadIdx.x];
        s += v.x + v.y + v.z + v.w;
    }
    red[threadIdx.x] = s;
    __syncthreads();
    for (int off = 128; off > 0; off >>= 1) {
        if (threadIdx.x < off) red[threadIdx.x] += red[threadIdx.x + off];
        __syncthreads();
    }
    if (threadIdx.x == 0) {
        float d = red[0];
        norm[row] = d > 0.f ? rsqrtf(d) : 0.f;
    }
}

// ---- pass 2: FnT[f][j] = bf16(norm[j] * F[j][f])  (tiled transpose) ----
__global__ void k_prep_fnt(const float* __restrict__ F, const float* __restrict__ norm,
                           unsigned short* __restrict__ FnT) {
    __shared__ unsigned short s[32][FF + 8];
    const int j0 = blockIdx.x * 32;
    #pragma unroll
    for (int i = 0; i < 16; ++i) {
        int idx = threadIdx.x + i * 256;       // 0..4095
        int j = idx >> 7, k = idx & 127;
        s[j][k] = f2bf(norm[j0 + j] * F[(size_t)(j0 + j) * FF + k]);
    }
    __syncthreads();
    #pragma unroll
    for (int i = 0; i < 16; ++i) {
        int idx = threadIdx.x + i * 256;
        int k = idx >> 5, jj = idx & 31;
        FnT[(size_t)k * NN + j0 + jj] = s[jj][k];
    }
}

// ---------------- W -> bf16 (row-major = transposed-B layout) ----------------
__global__ void k_prep_w(const float* __restrict__ W, unsigned short* __restrict__ Wb) {
    int i = blockIdx.x * 256 + threadIdx.x;
    Wb[i] = f2bf(W[i]);
}

// ---------------- main GEMM: K-split partial agg ----------------
__global__ __launch_bounds__(256) void k_gcn_partial(
    const float* __restrict__ A, const unsigned short* __restrict__ FnT,
    float* __restrict__ Pagg)
{
    __shared__ unsigned short bchunk[128 * SP];   // Fn chunk: [feature n][k within chunk]

    const int tid  = threadIdx.x;
    const int wave = tid >> 5;
    const int lane = tid & 31;
    const int l16  = lane & 15;
    const int hi   = lane >> 4;
    const int m0   = blockIdx.x * 128;            // WG row base
    const int wrow = wave * 16;                   // wave-local row base
    const int kbeg = blockIdx.y * KRANGE;         // this WG's K quarter

    f32x8 acc[8];
    #pragma unroll
    for (int nt = 0; nt < 8; ++nt)
        #pragma unroll
        for (int r = 0; r < 8; ++r) acc[nt][r] = 0.f;

    const size_t arow = (size_t)(m0 + wrow + l16) * NN;

    for (int k0 = kbeg; k0 < kbeg + KRANGE; k0 += KC) {
        __syncthreads();
        // cooperative stage of Fn chunk: FnT[n][k0..k0+KC) -> LDS
        {
            const int n    = tid >> 1;
            const int half = tid & 1;
            const uint4* g = (const uint4*)(FnT + (size_t)n * NN + k0 + half * 64);
            uint4* d       = (uint4*)(bchunk + n * SP + half * 64);
            #pragma unroll
            for (int i = 0; i < 8; ++i) d[i] = g[i];
        }
        __syncthreads();

        #pragma unroll            // unrolled so the chunk's 16 A-loads pipeline across steps
        for (int kk = 0; kk < KC; kk += 32) {
            // A fragment straight from global f32, converted to bf16.
            // A layout (16-bit, 16x32): lanes0-15 K=b..b+7 & b+16..b+23; lanes16-31 the +8 halves.
            BFrag a;
            {
                const int kb = k0 + kk + hi * 8;
                const float4* ap = (const float4*)(A + arow + kb);
                float4 x0 = ap[0], x1 = ap[1];     // K kb .. kb+7
                float4 x2 = ap[4], x3 = ap[5];     // K kb+16 .. kb+23
                a.u[0]=f2bf(x0.x);  a.u[1]=f2bf(x0.y);  a.u[2]=f2bf(x0.z);  a.u[3]=f2bf(x0.w);
                a.u[4]=f2bf(x1.x);  a.u[5]=f2bf(x1.y);  a.u[6]=f2bf(x1.z);  a.u[7]=f2bf(x1.w);
                a.u[8]=f2bf(x2.x);  a.u[9]=f2bf(x2.y);  a.u[10]=f2bf(x2.z); a.u[11]=f2bf(x2.w);
                a.u[12]=f2bf(x3.x); a.u[13]=f2bf(x3.y); a.u[14]=f2bf(x3.z); a.u[15]=f2bf(x3.w);
            }
            const int koff = kk + hi * 8;    // B layout: lane = column, K koff..koff+7 & koff+16..koff+23 per half-wave
            const unsigned short* bbase = bchunk + l16 * SP + koff;

            // depth-3 rotating B-fragment pipeline: frag nt and nt+1 stay live while
            // frag nt+2 is loading -> regalloc must keep them in distinct registers.
            BFrag bf[3];
            #pragma unroll
            for (int p = 0; p < 2; ++p) {
                const unsigned short* bp = bbase + p * 16 * SP;
                bf[p].q[0] = ((const uint4*)bp)[0];
                bf[p].q[1] = ((const uint4*)(bp + 16))[0];
            }
            #pragma unroll
            for (int nt = 0; nt < 8; ++nt) {
                if (nt + 2 < 8) {
                    const unsigned short* bp = bbase + (nt + 2) * 16 * SP;
                    bf[(nt + 2) % 3].q[0] = ((const uint4*)bp)[0];
                    bf[(nt + 2) % 3].q[1] = ((const uint4*)(bp + 16))[0];
                }
                acc[nt] = __builtin_amdgcn_wmma_f32_16x16x32_bf16(
                    false, a.v, false, bf[nt % 3].v, (short)0, acc[nt], false, false);
            }
            // pin issue order: 4 DS reads, then (1 WMMA, 2 DS reads) x6, then 2 WMMA
            SGB(0x100, 4, 0);
            #pragma unroll
            for (int nt = 0; nt < 6; ++nt) { SGB(0x008, 1, 0); SGB(0x100, 2, 0); }
            SGB(0x008, 2, 0);
        }
    }

    // store unscaled f32 partial: Pagg[split][row][f]
    float* P = Pagg + (size_t)blockIdx.y * NN * FF;
    #pragma unroll
    for (int nt = 0; nt < 8; ++nt) {
        float* op = P + (size_t)(m0 + wrow + hi * 8) * FF + nt * 16 + l16;
        #pragma unroll
        for (int r = 0; r < 8; ++r) op[(size_t)r * FF] = acc[nt][r];
    }
}

// ------- epilogue: sum partials, scale rows, agg @ W^T + b via WMMA -------
__global__ __launch_bounds__(256) void k_gcn_final(
    const float* __restrict__ Pagg, const float* __restrict__ norm,
    const unsigned short* __restrict__ Wb, const float* __restrict__ bias,
    float* __restrict__ out)
{
    __shared__ unsigned short aggs[128 * SP2];
    const int tid = threadIdx.x;
    const int m0  = blockIdx.x * 128;

    // sum the KSPLIT partials, scale by norm[row], deposit bf16 block in LDS
    #pragma unroll
    for (int i = 0; i < 16; ++i) {
        int idx = tid + i * 256;         // float4 index within 128x128 block
        int row = idx >> 5;              // 32 float4 per row
        int f4  = idx & 31;
        size_t g = (size_t)(m0 + row) * FF + f4 * 4;
        float4 s = make_float4(0.f, 0.f, 0.f, 0.f);
        #pragma unroll
        for (int sp = 0; sp < KSPLIT; ++sp) {
            float4 v = *(const float4*)(Pagg + (size_t)sp * NN * FF + g);
            s.x += v.x; s.y += v.y; s.z += v.z; s.w += v.w;
        }
        float rn = norm[m0 + row];
        union { unsigned short u[4]; uint2 d; } p;
        p.u[0] = f2bf(s.x * rn); p.u[1] = f2bf(s.y * rn);
        p.u[2] = f2bf(s.z * rn); p.u[3] = f2bf(s.w * rn);
        *(uint2*)&aggs[row * SP2 + f4 * 4] = p.d;
    }
    __syncthreads();

    const int wave = tid >> 5;
    const int lane = tid & 31;
    const int l16  = lane & 15;
    const int hi   = lane >> 4;
    const int wrow = wave * 16;

    f32x8 acc2[8];
    #pragma unroll
    for (int nt = 0; nt < 8; ++nt) {
        float bv = bias[nt * 16 + l16];
        #pragma unroll
        for (int r = 0; r < 8; ++r) acc2[nt][r] = bv;
    }
    #pragma unroll
    for (int k2 = 0; k2 < FF; k2 += 32) {
        BFrag a2;
        const int kb = k2 + hi * 8;
        const unsigned short* abase = aggs + (wrow + l16) * SP2 + kb;
        a2.q[0] = *(const uint4*)(abase);
        a2.q[1] = *(const uint4*)(abase + 16);
        #pragma unroll
        for (int nt = 0; nt < 8; ++nt) {
            BFrag b2;
            const uint4* bp2 = (const uint4*)(Wb + (nt * 16 + l16) * FF + k2 + hi * 8);
            b2.q[0] = bp2[0];
            b2.q[1] = bp2[2];
            acc2[nt] = __builtin_amdgcn_wmma_f32_16x16x32_bf16(
                false, a2.v, false, b2.v, (short)0, acc2[nt], false, false);
        }
    }

    #pragma unroll
    for (int nt = 0; nt < 8; ++nt) {
        float* op = out + (size_t)(m0 + wrow + hi * 8) * FF + nt * 16 + l16;
        #pragma unroll
        for (int r = 0; r < 8; ++r) op[(size_t)r * FF] = acc2[nt][r];
    }
}

extern "C" void kernel_launch(void* const* d_in, const int* in_sizes, int n_in,
                              void* d_out, int out_size, void* d_ws, size_t ws_size,
                              hipStream_t stream) {
    (void)in_sizes; (void)n_in; (void)out_size; (void)ws_size;
    const float* A = (const float*)d_in[0];
    const float* F = (const float*)d_in[1];
    const float* W = (const float*)d_in[2];
    const float* b = (const float*)d_in[3];
    float* out = (float*)d_out;

    // workspace: norm (64 KB) | FnT bf16 (4 MiB) | W bf16 (32 KB) | partials (32 MiB)
    char* ws = (char*)d_ws;
    float* norm = (float*)ws;
    unsigned short* FnT = (unsigned short*)(ws + 65536);
    unsigned short* Wb  = (unsigned short*)(ws + 65536 + (size_t)FF * NN * 2);
    float* Pagg = (float*)(ws + 65536 + (size_t)FF * NN * 2 + (size_t)FF * FF * 2);

    k_rowsum   <<<NN, 256, 0, stream>>>(A, norm);
    k_prep_fnt <<<NN / 32, 256, 0, stream>>>(F, norm, FnT);
    k_prep_w   <<<(FF * FF) / 256, 256, 0, stream>>>(W, Wb);
    k_gcn_partial<<<dim3(NN / 128, KSPLIT), 256, 0, stream>>>(A, FnT, Pagg);
    k_gcn_final  <<<NN / 128, 256, 0, stream>>>(Pagg, norm, Wb, b, out);
}
